// ActorNetwork_45337674777312
// MI455X (gfx1250) — matrix operand order repaired and checked by reading, not compiled, not run on hardware
//
#include <hip/hip_runtime.h>

typedef __attribute__((ext_vector_type(16))) _Float16 v16h;
typedef __attribute__((ext_vector_type(8)))  float    v8f;

#define LSTR 72          // LDS tile row stride in halves: 144B rows -> 16B-aligned, conflict-free
#define WPB  4           // waves per block (wave32)
#define TPB  (WPB * 32)  // 128 threads

struct MlpG { const unsigned int *P0, *P1, *P2; const float *b0, *b1, *b2; };              // 3 WMMA layers
struct MlpH { const unsigned int *P0, *P1; const float *b0, *b1, *W2, *b2; };              // 2 WMMA + scalar head

// One wave computes a 16-row layer: out[16][MOUT] = act(in[16][KINpad] @ W + b).
// A: two ds_load_b128 per k-tile (ISA 7.12.2 16-bit A layout, tile K-padded to 32*KT).
// B: two coalesced 16B loads from the pre-packed fragment table (already f16, zero-padded).
// OPAD: zero-fill cols [MOUT, OPAD) so the next layer can use unconditional vector loads.
template <int KIN, int MOUT, bool RELU, int OPAD>
__device__ __forceinline__ void wmma_layer(const unsigned int* __restrict__ Wp,
                                           const float* __restrict__ bias,
                                           const _Float16* in, _Float16* out)
{
  constexpr int KT = (KIN + 31) / 32;
  constexpr int NT = MOUT / 16;
  const int lane = threadIdx.x & 31;
  const int hi   = lane >> 4;
  const int l16  = lane & 15;
#pragma unroll
  for (int nt = 0; nt < NT; ++nt) {
    const int col = nt * 16 + l16;
    const float bv = bias[col];
    v8f c;
#pragma unroll
    for (int v = 0; v < 8; ++v) c[v] = bv;
#pragma unroll
    for (int kt = 0; kt < KT; ++kt) {
      const _Float16* ap = in + l16 * LSTR + kt * 32 + hi * 8;
      union { uint4 q[2]; v16h h; } ua, ub;
      ua.q[0] = *(const uint4*)(ap);        // K = kt*32 + hi*8 .. +7
      ua.q[1] = *(const uint4*)(ap + 16);   // K = kt*32 + 16 + hi*8 .. +7
      const unsigned int* bp = Wp + ((kt * NT + nt) * 32 + lane) * 8;
      ub.q[0] = *(const uint4*)(bp);
      ub.q[1] = *(const uint4*)(bp + 4);
      c = __builtin_amdgcn_wmma_f32_16x16x32_f16(false, ua.h, false, ub.h, (short)0, c, false, false);
    }
#pragma unroll
    for (int v = 0; v < 8; ++v) {
      float val = c[v];
      if (RELU) val = fmaxf(val, 0.0f);
      out[(v + hi * 8) * LSTR + col] = (_Float16)val;  // D: VGPR v -> rows v / v+8
    }
  }
  if constexpr (OPAD > MOUT) {
    constexpr int ZW2 = (OPAD - MOUT) / 2;  // dword columns to zero
    for (int idx = lane; idx < 16 * ZW2; idx += 32) {
      const int row = idx / ZW2;
      const int c2  = (idx % ZW2) * 2;
      *(unsigned int*)(out + row * LSTR + MOUT + c2) = 0u;
    }
  }
}

// Final 16 -> 1 head: scalar dot per row (WMMA would waste 15/16 columns).
__device__ __forceinline__ float head16(const float* __restrict__ W,
                                        const float* __restrict__ bias,
                                        const _Float16* act, int row)
{
  float s = bias[0];
#pragma unroll
  for (int k = 0; k < 16; ++k) s += (float)act[row * LSTR + k] * W[k];
  return s;
}

// ---- weight pre-pack: f32 W[KIN][MOUT] -> B-fragment table (f16 pairs) ------
__global__ void k_pack(const float* __restrict__ W, unsigned int* __restrict__ out,
                       int KIN, int MOUT)
{
  const int KT = (KIN + 31) / 32, NT = MOUT / 16;
  const int total = KT * NT * 256;
  for (int i = blockIdx.x * blockDim.x + threadIdx.x; i < total; i += gridDim.x * blockDim.x) {
    const int v = i & 7, lane = (i >> 3) & 31, t = i >> 8;
    const int nt = t % NT, kt = t / NT;
    const int kb  = kt * 32 + (lane >> 4) * 8 + 2 * (v & 3) + (v >> 2) * 16;
    const int col = nt * 16 + (lane & 15);
    const _Float16 lo = (kb     < KIN) ? (_Float16)W[(size_t)kb * MOUT + col]       : (_Float16)0.0f;
    const _Float16 hi = (kb + 1 < KIN) ? (_Float16)W[(size_t)(kb + 1) * MOUT + col] : (_Float16)0.0f;
    const unsigned short ul = __builtin_bit_cast(unsigned short, lo);
    const unsigned short uh = __builtin_bit_cast(unsigned short, hi);
    out[i] = (unsigned int)ul | ((unsigned int)uh << 16);
  }
}

// ---- meta: obs_indptr, dag_obs, int outputs --------------------------------
__global__ void k_meta(const int* __restrict__ ptr, const int* __restrict__ ndpo,
                       int* __restrict__ obs_ip, int* __restrict__ dag_obs,
                       float* __restrict__ out_tail, int G, int B)
{
  __shared__ int ip[64];
  if (threadIdx.x == 0) {
    int acc = 0; ip[0] = 0;
    for (int i = 0; i < B; ++i) { acc += ndpo[i]; ip[i + 1] = acc; }
  }
  __syncthreads();
  if ((int)threadIdx.x <= B) {
    obs_ip[threadIdx.x] = ip[threadIdx.x];
    out_tail[B + threadIdx.x] = (float)ip[threadIdx.x];                 // obs_indptr
  }
  if ((int)threadIdx.x < B)                                            // telescoped segment_sum
    out_tail[threadIdx.x] = (float)(ptr[ip[threadIdx.x + 1]] - ptr[ip[threadIdx.x]]);
  for (int g = threadIdx.x; g < G; g += blockDim.x) {
    int o = 0;
    while (o < B - 1 && g >= ip[o + 1]) ++o;
    dag_obs[g] = o;
  }
}

__global__ void k_zero(float* __restrict__ p, long n)
{
  long i = (long)blockIdx.x * blockDim.x + threadIdx.x;
  if (i < n) p[i] = 0.0f;
}

// ---- prep MLP: x(N,6) -> h(N,16) -------------------------------------------
__global__ void __launch_bounds__(TPB)
k_prep(const float* __restrict__ x, float* __restrict__ h, MlpG p, int N)
{
  __shared__ __align__(16) _Float16 sm[WPB][2][16 * LSTR];
  const int wave = threadIdx.x >> 5, lane = threadIdx.x & 31;
  _Float16* tA = sm[wave][0];
  _Float16* tB = sm[wave][1];
  const long r0 = ((long)blockIdx.x * WPB + wave) * 16;
  for (int idx = lane; idx < 16 * 32; idx += 32) {          // K-padded to 32
    const int row = idx >> 5, c = idx & 31;
    const long n = r0 + row;
    tA[row * LSTR + c] = (c < 6 && n < N) ? (_Float16)x[n * 6 + c] : (_Float16)0.0f;
  }
  wmma_layer<6, 32, true, 32>(p.P0, p.b0, tA, tB);
  wmma_layer<32, 16, true, 32>(p.P1, p.b1, tB, tA);
  wmma_layer<16, 16, false, 16>(p.P2, p.b2, tA, tB);
  for (int idx = lane; idx < 256; idx += 32) {
    const int row = idx >> 4, c = idx & 15;
    const long n = r0 + row;
    if (n < N) h[n * 16 + c] = (float)tB[row * LSTR + c];
  }
}

// ---- edge scatter: aggr[dst] += h[src] -------------------------------------
__global__ void k_scatter(const int* __restrict__ ei, const float* __restrict__ h,
                          float* __restrict__ aggr, long E)
{
  const long e = (long)blockIdx.x * blockDim.x + threadIdx.x;
  if (e >= E) return;
  const long s = ei[e], d = ei[E + e];
  const float4* hv = (const float4*)(h + s * 16);
  float4 a0 = hv[0], a1 = hv[1], a2 = hv[2], a3 = hv[3];
  float* ap = aggr + d * 16;
  atomicAdd(ap +  0, a0.x); atomicAdd(ap +  1, a0.y); atomicAdd(ap +  2, a0.z); atomicAdd(ap +  3, a0.w);
  atomicAdd(ap +  4, a1.x); atomicAdd(ap +  5, a1.y); atomicAdd(ap +  6, a1.z); atomicAdd(ap +  7, a1.w);
  atomicAdd(ap +  8, a2.x); atomicAdd(ap +  9, a2.y); atomicAdd(ap + 10, a2.z); atomicAdd(ap + 11, a2.w);
  atomicAdd(ap + 12, a3.x); atomicAdd(ap + 13, a3.y); atomicAdd(ap + 14, a3.z); atomicAdd(ap + 15, a3.w);
}

// ---- fused agg -> proc -> node chain; one pass over N ----------------------
__global__ void __launch_bounds__(TPB)
k_node(const float* __restrict__ x, const float* __restrict__ aggr,
       const int* __restrict__ batch, float* __restrict__ nemb,
       float* __restrict__ dagemb, MlpG pa, MlpG pp, MlpG pn, int N)
{
  __shared__ __align__(16) _Float16 sm[WPB][2][16 * LSTR];
  const int wave = threadIdx.x >> 5, lane = threadIdx.x & 31;
  _Float16* tA = sm[wave][0];
  _Float16* tB = sm[wave][1];
  const long r0 = ((long)blockIdx.x * WPB + wave) * 16;

  float xr0 = 0, xr1 = 0, xr2 = 0, xr3 = 0, xr4 = 0, xr5 = 0;
  {
    const long n = r0 + (lane & 15);
    if (lane < 16 && n < N) {
      xr0 = x[n * 6 + 0]; xr1 = x[n * 6 + 1]; xr2 = x[n * 6 + 2];
      xr3 = x[n * 6 + 3]; xr4 = x[n * 6 + 4]; xr5 = x[n * 6 + 5];
    }
  }
  for (int idx = lane; idx < 16 * 32; idx += 32) {          // aggr tile, K-padded
    const int row = idx >> 5, c = idx & 31;
    const long n = r0 + row;
    tA[row * LSTR + c] = (c < 16 && n < N) ? (_Float16)aggr[n * 16 + c] : (_Float16)0.0f;
  }
  wmma_layer<16, 32, true, 32>(pa.P0, pa.b0, tA, tB);
  wmma_layer<32, 16, true, 32>(pa.P1, pa.b1, tB, tA);       // tA cols16-31 zeroed
  wmma_layer<16, 16, false, 16>(pa.P2, pa.b2, tA, tB);      // x_agg in tB

  if (lane < 16) {                                          // concat [x, x_agg]; cols22-31 stay 0
    _Float16* r = tA + (lane & 15) * LSTR;
    r[0] = (_Float16)xr0; r[1] = (_Float16)xr1; r[2] = (_Float16)xr2;
    r[3] = (_Float16)xr3; r[4] = (_Float16)xr4; r[5] = (_Float16)xr5;
  }
  for (int idx = lane; idx < 256; idx += 32) {
    const int row = idx >> 4, c = idx & 15;
    tA[row * LSTR + 6 + c] = tB[row * LSTR + c];
  }
  wmma_layer<22, 32, true, 32>(pp.P0, pp.b0, tA, tB);
  wmma_layer<32, 16, true, 32>(pp.P1, pp.b1, tB, tA);
  wmma_layer<16, 16, false, 16>(pp.P2, pp.b2, tA, tB);      // node_embeddings in tB
  for (int idx = lane; idx < 256; idx += 32) {
    const int row = idx >> 4, c = idx & 15;
    const long n = r0 + row;
    if (n < N) nemb[n * 16 + c] = (float)tB[row * LSTR + c];
  }
  if (lane < 16) {                                          // concat [x, node_emb]
    _Float16* r = tA + (lane & 15) * LSTR;
    r[0] = (_Float16)xr0; r[1] = (_Float16)xr1; r[2] = (_Float16)xr2;
    r[3] = (_Float16)xr3; r[4] = (_Float16)xr4; r[5] = (_Float16)xr5;
  }
  for (int idx = lane; idx < 256; idx += 32) {
    const int row = idx >> 4, c = idx & 15;
    tA[row * LSTR + 6 + c] = tB[row * LSTR + c];
  }
  wmma_layer<22, 32, true, 32>(pn.P0, pn.b0, tA, tB);
  wmma_layer<32, 16, true, 32>(pn.P1, pn.b1, tB, tA);
  wmma_layer<16, 16, false, 16>(pn.P2, pn.b2, tA, tB);      // nodes_merged in tB
  for (int idx = lane; idx < 256; idx += 32) {
    const int row = idx >> 4, c = idx & 15;
    const long n = r0 + row;
    if (n < N) atomicAdd(&dagemb[(long)batch[n] * 16 + c], (float)tB[row * LSTR + c]);
  }
}

// ---- dag MLP: dagemb(G,16) -> dag_z; accumulate per-obs --------------------
__global__ void __launch_bounds__(TPB)
k_dag(const float* __restrict__ dagemb, float* __restrict__ glob,
      const int* __restrict__ dag_obs, MlpG p, int G)
{
  __shared__ __align__(16) _Float16 sm[WPB][2][16 * LSTR];
  const int wave = threadIdx.x >> 5, lane = threadIdx.x & 31;
  _Float16* tA = sm[wave][0];
  _Float16* tB = sm[wave][1];
  const long r0 = ((long)blockIdx.x * WPB + wave) * 16;
  for (int idx = lane; idx < 16 * 32; idx += 32) {
    const int row = idx >> 5, c = idx & 31;
    const long g = r0 + row;
    tA[row * LSTR + c] = (c < 16 && g < G) ? (_Float16)dagemb[g * 16 + c] : (_Float16)0.0f;
  }
  wmma_layer<16, 32, true, 32>(p.P0, p.b0, tA, tB);
  wmma_layer<32, 16, true, 32>(p.P1, p.b1, tB, tA);
  wmma_layer<16, 16, false, 16>(p.P2, p.b2, tA, tB);
  for (int idx = lane; idx < 256; idx += 32) {
    const int row = idx >> 4, c = idx & 15;
    const long g = r0 + row;
    if (g < G) atomicAdd(&glob[(long)dag_obs[g] * 16 + c], (float)tB[row * LSTR + c]);
  }
}

__global__ void k_div(float* __restrict__ glob, const int* __restrict__ ndpo, int B)
{
  int i = blockIdx.x * blockDim.x + threadIdx.x;
  if (i < B * 16) glob[i] /= (float)ndpo[i >> 4];
}

// ---- node score: concat(54, pad 64) -> 32 -> 16 -> 1 -----------------------
__global__ void __launch_bounds__(TPB)
k_nscore(const float* __restrict__ x, const float* __restrict__ nemb,
         const float* __restrict__ dagemb, const float* __restrict__ glob,
         const int* __restrict__ batch, const int* __restrict__ dag_obs,
         float* __restrict__ out, MlpH p, int N)
{
  __shared__ __align__(16) _Float16 sm[WPB][2][16 * LSTR];
  __shared__ int sg[WPB][16], so[WPB][16];
  const int wave = threadIdx.x >> 5, lane = threadIdx.x & 31;
  _Float16* tA = sm[wave][0];
  _Float16* tB = sm[wave][1];
  const long r0 = ((long)blockIdx.x * WPB + wave) * 16;
  if (lane < 16) {
    const long n = r0 + lane;
    const int g = (n < N) ? batch[n] : 0;
    sg[wave][lane] = g;
    so[wave][lane] = dag_obs[g];
  }
  for (int idx = lane; idx < 16 * 64; idx += 32) {
    const int row = idx >> 6, c = idx & 63;
    const long n = r0 + row;
    float v = 0.0f;
    if (c < 54 && n < N) {
      const int g = sg[wave][row], o = so[wave][row];
      v = (c < 6)  ? x[n * 6 + c]
        : (c < 22) ? nemb[n * 16 + (c - 6)]
        : (c < 38) ? dagemb[(long)g * 16 + (c - 22)]
                   : glob[(long)o * 16 + (c - 38)];
    }
    tA[row * LSTR + c] = (_Float16)v;
  }
  wmma_layer<54, 32, true, 32>(p.P0, p.b0, tA, tB);
  wmma_layer<32, 16, true, 16>(p.P1, p.b1, tB, tA);
  const float s = head16(p.W2, p.b2, tA, lane & 15);
  const long n = r0 + (lane & 15);
  if (lane < 16 && n < N) out[n] = s;
}

// ---- dag score: (G*W rows, 36 feats pad 64) -> 32 -> 16 -> 1 ---------------
__global__ void __launch_bounds__(TPB)
k_dscore(const float* __restrict__ x, const float* __restrict__ dagemb,
         const float* __restrict__ glob, const int* __restrict__ ptr,
         const int* __restrict__ dag_obs, float* __restrict__ out,
         MlpH p, int Wk, long R)
{
  __shared__ __align__(16) _Float16 sm[WPB][2][16 * LSTR];
  __shared__ int sg[WPB][16], so[WPB][16];
  const int wave = threadIdx.x >> 5, lane = threadIdx.x & 31;
  _Float16* tA = sm[wave][0];
  _Float16* tB = sm[wave][1];
  const long r0 = ((long)blockIdx.x * WPB + wave) * 16;
  if (lane < 16) {
    const long r = r0 + lane;
    const int g = (r < R) ? (int)(r / Wk) : 0;
    sg[wave][lane] = g;
    so[wave][lane] = dag_obs[g];
  }
  for (int idx = lane; idx < 16 * 64; idx += 32) {
    const int row = idx >> 6, c = idx & 63;
    const long r = r0 + row;
    float v = 0.0f;
    if (c < 36 && r < R) {
      const int g = sg[wave][row];
      if (c < 3)       v = x[(long)ptr[g] * 6 + c];
      else if (c < 19) v = dagemb[(long)g * 16 + (c - 3)];
      else if (c < 35) v = glob[(long)so[wave][row] * 16 + (c - 19)];
      else             v = (float)(r - (long)g * Wk);
    }
    tA[row * LSTR + c] = (_Float16)v;
  }
  wmma_layer<36, 32, true, 32>(p.P0, p.b0, tA, tB);
  wmma_layer<32, 16, true, 16>(p.P1, p.b1, tB, tA);
  const float s = head16(p.W2, p.b2, tA, lane & 15);
  const long r = r0 + (lane & 15);
  if (lane < 16 && r < R) out[r] = s;
}

extern "C" void kernel_launch(void* const* d_in, const int* in_sizes, int n_in,
                              void* d_out, int out_size, void* d_ws, size_t ws_size,
                              hipStream_t stream)
{
  const float* x     = (const float*)d_in[0];
  const int*   ei    = (const int*)d_in[1];
  const int*   batch = (const int*)d_in[2];
  const int*   ptr   = (const int*)d_in[3];
  const int*   ndpo  = (const int*)d_in[4];
  const int  N  = in_sizes[0] / 6;
  const long E  = (long)in_sizes[1] / 2;
  const int  G  = in_sizes[3] - 1;
  const int  B  = in_sizes[4];
  const int  GW = out_size - N - (2 * B + 1);
  const int  Wk = (G > 0 && GW > 0) ? GW / G : 1;

  // Locate MLP params by flat size (skips PRNG-key leaves in the pytree).
  static const int LDIMS[7][4] = {
    {6, 32, 16, 16},  // prep
    {22, 32, 16, 16}, // proc
    {16, 32, 16, 16}, // agg
    {22, 32, 16, 16}, // node
    {16, 32, 16, 16}, // dag
    {54, 32, 16, 1},  // node_score
    {36, 32, 16, 1},  // dag_score
  };
  const float* w[7][3]; const float* b[7][3];
  {
    int idx = 5;
    for (int m = 0; m < 7; ++m)
      for (int l = 0; l < 3; ++l) {
        const int wsz = LDIMS[m][l] * LDIMS[m][l + 1];
        while (idx < n_in && in_sizes[idx] != wsz) ++idx;
        w[m][l] = (const float*)d_in[idx < n_in ? idx : n_in - 1];
        if (idx < n_in) ++idx;
        const int bsz = LDIMS[m][l + 1];
        while (idx < n_in && in_sizes[idx] != bsz) ++idx;
        b[m][l] = (const float*)d_in[idx < n_in ? idx : n_in - 1];
        if (idx < n_in) ++idx;
      }
  }

  // Workspace layout (floats): h/nemb | aggr | dagemb | glob | ints | packed weights
  float* ws      = (float*)d_ws;
  float* h       = ws;                               // N*16, reused as nemb
  float* aggr    = ws + (size_t)N * 16;              // N*16
  float* dagemb  = aggr + (size_t)N * 16;            // G*16
  float* glob    = dagemb + (size_t)G * 16;          // B*16
  int*   dag_obs = (int*)(glob + (size_t)B * 16);    // G
  int*   obs_ip  = dag_obs + G;                      // B+1
  uintptr_t pa = ((uintptr_t)(obs_ip + (B + 1)) + 15) & ~(uintptr_t)15;
  unsigned int* pkbase = (unsigned int*)pa;

  // Pre-pack WMMA layers (all but the two 16->1 heads) into B-fragment tables.
  unsigned int* pk[7][3];
  {
    size_t off = 0;
    for (int m = 0; m < 7; ++m)
      for (int l = 0; l < 3; ++l) {
        if (m >= 5 && l == 2) { pk[m][l] = nullptr; continue; }
        const int K = LDIMS[m][l], M = LDIMS[m][l + 1];
        const size_t sz = (size_t)((K + 31) / 32) * (M / 16) * 256;
        pk[m][l] = pkbase + off;
        off += sz;
        k_pack<<<8, 256, 0, stream>>>(w[m][l], pk[m][l], K, M);
      }
  }
  MlpG P[5];
  for (int m = 0; m < 5; ++m)
    P[m] = MlpG{pk[m][0], pk[m][1], pk[m][2], b[m][0], b[m][1], b[m][2]};
  MlpH Pns{pk[5][0], pk[5][1], b[5][0], b[5][1], w[5][2], b[5][2]};
  MlpH Pds{pk[6][0], pk[6][1], b[6][0], b[6][1], w[6][2], b[6][2]};

  float* out        = (float*)d_out;
  float* out_nscore = out;
  float* out_dscore = out + N;
  float* out_tail   = out + N + GW;

  k_meta<<<1, 128, 0, stream>>>(ptr, ndpo, obs_ip, dag_obs, out_tail, G, B);

  const long nz = (long)N * 16 + (long)G * 16 + (long)B * 16;
  k_zero<<<(unsigned)((nz + 255) / 256), 256, 0, stream>>>(aggr, nz);

  const unsigned nb = (unsigned)((N + WPB * 16 - 1) / (WPB * 16));
  k_prep<<<nb, TPB, 0, stream>>>(x, h, P[0], N);

  k_scatter<<<(unsigned)((E + 255) / 256), 256, 0, stream>>>(ei, h, aggr, E);

  k_node<<<nb, TPB, 0, stream>>>(x, aggr, batch, h /*nemb*/, dagemb, P[2], P[1], P[3], N);

  k_dag<<<(unsigned)((G + WPB * 16 - 1) / (WPB * 16)), TPB, 0, stream>>>(dagemb, glob, dag_obs, P[4], G);

  k_div<<<(unsigned)((B * 16 + 127) / 128), 128, 0, stream>>>(glob, ndpo, B);

  k_nscore<<<nb, TPB, 0, stream>>>(x, h, dagemb, glob, batch, dag_obs, out_nscore, Pns, N);

  k_dscore<<<(unsigned)(((long)GW + WPB * 16 - 1) / (WPB * 16)), TPB, 0, stream>>>(
      x, dagemb, glob, ptr, dag_obs, out_dscore, Pds, Wk, (long)GW);
}
